// AttentionPooling_68358699483266
// MI455X (gfx1250) — matrix operand order repaired and checked by reading, not compiled, not run on hardware
//
#include <hip/hip_runtime.h>
#include <hip/hip_bf16.h>

typedef __attribute__((ext_vector_type(16))) _Float16 v16h;
typedef __attribute__((ext_vector_type(8)))  _Float16 v8h;
typedef __attribute__((ext_vector_type(4)))  _Float16 v4h;
typedef __attribute__((ext_vector_type(8)))  float    v8f;

#define D_IN  256
#define D_ATT 128
#define ROWS  128          // rows (nodes) per workgroup
#define LDW   (D_IN + 8)   // padded LDS row: 264 f16 = 528B -> 16B bank shift per row

// ---- prep: W1 [256,128] f32 row-major -> W1^T [128,256] f16 row-major -------
__global__ void w1_transpose_f16(const float* __restrict__ W1,
                                 _Float16* __restrict__ w1t) {
  int n = blockIdx.x;   // 0..127 (output column of W1)
  int k = threadIdx.x;  // 0..255 (input feature)
  w1t[n * D_IN + k] = (_Float16)W1[k * D_ATT + n];
}

// ---- zero the output accumulator (harness poisons d_out) --------------------
__global__ void zero_f32(float* __restrict__ p) {
  p[(size_t)blockIdx.x * blockDim.x + threadIdx.x] = 0.0f;
}

// ---- fused attention-pool kernel -------------------------------------------
__global__ __launch_bounds__(256)
void attn_pool_kernel(const float* __restrict__ x,
                      const int* __restrict__ batch,
                      const _Float16* __restrict__ w1t_g,
                      const float* __restrict__ b1,
                      const float* __restrict__ W2,
                      const float* __restrict__ b2,
                      float* __restrict__ out,
                      int n_nodes) {
  __shared__ alignas(16) _Float16 xs[ROWS][LDW];    // x tile, f16
  __shared__ alignas(16) _Float16 w1s[D_ATT][LDW];  // W1^T, f16
  __shared__ float attn_s[ROWS];
  __shared__ int   batch_s[ROWS];
  __shared__ float w2_s[D_ATT];
  __shared__ float b1_s[D_ATT];
  __shared__ float b2_s;

  const int tid  = threadIdx.x;
  const int row0 = blockIdx.x * ROWS;

  // Phase 0a: W1^T f16 -> LDS (16B per thread per iteration, coalesced)
  for (int i = tid; i < D_ATT * D_IN / 8; i += 256) {
    int n = i >> 5;           // 32 chunks of 8 f16 per row
    int k = (i & 31) << 3;
    *(v8h*)&w1s[n][k] = *(const v8h*)&w1t_g[n * D_IN + k];
  }
  // Phase 0b: x tile f32 -> f16 LDS (float4 coalesced loads, zero-pad tail rows)
  for (int i = tid; i < ROWS * D_IN / 4; i += 256) {
    int r  = i >> 6;
    int k4 = (i & 63) << 2;
    float4 v = make_float4(0.f, 0.f, 0.f, 0.f);
    int row = row0 + r;
    if (row < n_nodes)
      v = *(const float4*)&x[(size_t)row * D_IN + k4];
    v4h h;
    h[0] = (_Float16)v.x; h[1] = (_Float16)v.y;
    h[2] = (_Float16)v.z; h[3] = (_Float16)v.w;
    *(v4h*)&xs[r][k4] = h;
  }
  if (tid < D_ATT) { w2_s[tid] = W2[tid]; b1_s[tid] = b1[tid]; }
  if (tid < ROWS) {
    int row = row0 + tid;
    batch_s[tid] = (row < n_nodes) ? batch[row] : 0;
  }
  if (tid == 0) b2_s = b2[0];
  __syncthreads();

  // Phase 1: each wave computes h_pre = x_tile(16x256) @ W1(256x128) via WMMA
  const int wave = tid >> 5;
  const int lane = tid & 31;
  const int hf   = lane >> 4;   // lane-group half (K split)
  const int l    = lane & 15;   // M (A) / N (B) index within tile
  const int arow = wave * 16 + l;

  v8f acc[8];
  {
    v8f z = {0.f, 0.f, 0.f, 0.f, 0.f, 0.f, 0.f, 0.f};
    #pragma unroll
    for (int nt = 0; nt < 8; ++nt) acc[nt] = z;
  }

  #pragma unroll
  for (int kt = 0; kt < 8; ++kt) {
    const int k0 = kt * 32;
    // A fragment (16x32 f16): V0-3 = K[k0 + hf*8 .. +7], V4-7 = K[k0+16+hf*8 .. +7]
    v8h alo = *(const v8h*)&xs[arow][k0 + hf * 8];
    v8h ahi = *(const v8h*)&xs[arow][k0 + 16 + hf * 8];
    v16h a;
    #pragma unroll
    for (int j = 0; j < 8; ++j) { a[j] = alo[j]; a[j + 8] = ahi[j]; }
    #pragma unroll
    for (int nt = 0; nt < 8; ++nt) {
      const int n = nt * 16 + l;
      // B fragment (32x16 f16): lane holds col n, K[k0 + hf*8 .. +7], K[k0+16+hf*8 .. +7]
      v8h blo = *(const v8h*)&w1s[n][k0 + hf * 8];
      v8h bhi = *(const v8h*)&w1s[n][k0 + 16 + hf * 8];
      v16h b;
      #pragma unroll
      for (int j = 0; j < 8; ++j) { b[j] = blo[j]; b[j + 8] = bhi[j]; }
      acc[nt] = __builtin_amdgcn_wmma_f32_16x16x32_f16(
          false, a, false, b, (short)0, acc[nt], false, false);
    }
  }

  // Epilogue: attn partial = sum_n tanh(h_pre + b1[n]) * W2[n]
  // C/D layout: lane l holds col n=nt*16+l, rows M = r + hf*8 (r = vgpr idx)
  float partial[8];
  #pragma unroll
  for (int r = 0; r < 8; ++r) partial[r] = 0.0f;
  #pragma unroll
  for (int nt = 0; nt < 8; ++nt) {
    const int n = nt * 16 + l;
    const float w2v = w2_s[n];
    const float b1v = b1_s[n];
    #pragma unroll
    for (int r = 0; r < 8; ++r)
      partial[r] += tanhf(acc[nt][r] + b1v) * w2v;
  }
  // reduce over the 16 lanes of each half (masks < 16 stay within the half)
  #pragma unroll
  for (int mask = 1; mask <= 8; mask <<= 1) {
    #pragma unroll
    for (int r = 0; r < 8; ++r)
      partial[r] += __shfl_xor(partial[r], mask, 32);
  }
  if (l == 0) {
    #pragma unroll
    for (int r = 0; r < 8; ++r)
      attn_s[wave * 16 + hf * 8 + r] = partial[r] + b2_s;
  }
  __syncthreads();

  // Phase 2: weighted segment-sum; thread = output column; batch is sorted so
  // accumulate runs and emit one f32 atomic per (segment-run, column).
  const int c = tid;
  const int rows_here = min(ROWS, n_nodes - row0);
  float sum = 0.0f;
  int cur = -1;
  for (int r = 0; r < rows_here; ++r) {
    int g = batch_s[r];                       // uniform across threads
    if (g != cur) {
      if (cur >= 0) atomicAdd(&out[(size_t)cur * D_IN + c], sum);
      cur = g; sum = 0.0f;
    }
    // x re-read hits L2 (just fetched in phase 0b); keeps f32 precision
    sum += x[(size_t)(row0 + r) * D_IN + c] * attn_s[r];
  }
  if (cur >= 0) atomicAdd(&out[(size_t)cur * D_IN + c], sum);
}

// ---- host entry -------------------------------------------------------------
extern "C" void kernel_launch(void* const* d_in, const int* in_sizes, int n_in,
                              void* d_out, int out_size, void* d_ws, size_t ws_size,
                              hipStream_t stream) {
  const float* x     = (const float*)d_in[0];
  const int*   batch = (const int*)d_in[1];
  // d_in[2] = num_graphs (device scalar, unused: out_size gives it)
  const float* W1    = (const float*)d_in[3];
  const float* b1    = (const float*)d_in[4];
  const float* W2    = (const float*)d_in[5];
  const float* b2    = (const float*)d_in[6];
  float*       out   = (float*)d_out;
  const int n_nodes  = in_sizes[1];

  _Float16* w1t = (_Float16*)d_ws;  // 128*256*2 = 64 KB scratch

  w1_transpose_f16<<<dim3(D_ATT), dim3(D_IN), 0, stream>>>(W1, w1t);
  zero_f32<<<dim3(out_size / 256), dim3(256), 0, stream>>>(out);

  const int nblocks = (n_nodes + ROWS - 1) / ROWS;
  attn_pool_kernel<<<dim3(nblocks), dim3(256), 0, stream>>>(
      x, batch, w1t, b1, W2, b2, out, n_nodes);
}
